// DemandRouter_41334765256787
// MI455X (gfx1250) — compile-verified, matrix operand order checked by reading, not run on hardware
//
#include <hip/hip_runtime.h>

typedef __attribute__((ext_vector_type(16))) _Float16 v16h;
typedef __attribute__((ext_vector_type(8)))  float    v8f;
typedef __attribute__((ext_vector_type(4)))  float    v4f;
typedef __attribute__((ext_vector_type(4)))  unsigned int v4u;
typedef __attribute__((ext_vector_type(8)))  int      v8i;
typedef __attribute__((ext_vector_type(4)))  int      v4i;

#define B_   2
#define T_   4096
#define D_   1024
#define KQ_  128
#define TOPK 8

// LDS key-row stride in halfs: 128 data + 8 pad (TDM pads 4 DWORDs per 64-DWORD
// row) -> 68-word stride -> conflict-free ds_load_b128 across a half-wave.
#define KSTRIDE 136

// ---------------------------------------------------------------------------
// Convert x (fp32) -> f16, 4 elements per thread
// ---------------------------------------------------------------------------
__global__ __launch_bounds__(256) void cvt_x_kernel(const float* __restrict__ x,
                                                    _Float16* __restrict__ xh) {
    size_t i = ((size_t)blockIdx.x * 256 + threadIdx.x) * 4;
    #pragma unroll
    for (int j = 0; j < 4; ++j) xh[i + j] = (_Float16)x[i + j];
}

// ---------------------------------------------------------------------------
// Convert + transpose W (D x KQ, row-major) -> Wt (KQ x D, f16)
// ---------------------------------------------------------------------------
__global__ __launch_bounds__(256) void cvt_w_kernel(const float* __restrict__ w,
                                                    _Float16* __restrict__ wt) {
    int i = blockIdx.x * 256 + threadIdx.x;       // i < D_*KQ_
    int d = i / KQ_;
    int n = i % KQ_;
    wt[(size_t)n * D_ + d] = (_Float16)w[i];
}

// ---------------------------------------------------------------------------
// Gate: sigmoid(x . Wg + bg), fp32, one row per wave (8 rows / block)
// ---------------------------------------------------------------------------
__global__ __launch_bounds__(256) void gate_kernel(const float* __restrict__ x,
                                                   const float* __restrict__ Wg,
                                                   const float* __restrict__ bg,
                                                   float* __restrict__ gates) {
    int row  = blockIdx.x * 8 + (threadIdx.x >> 5);
    int lane = threadIdx.x & 31;
    const float* xr = x + (size_t)row * D_;
    float s = 0.0f;
    #pragma unroll 8
    for (int i = lane; i < D_; i += 32) s += xr[i] * Wg[i];
    #pragma unroll
    for (int off = 16; off > 0; off >>= 1) s += __shfl_xor(s, off, 32);
    if (lane == 0) gates[row] = 1.0f / (1.0f + __expf(-(s + bg[0])));
}

// ---------------------------------------------------------------------------
// Projection: q = (x@Wq + bq) * 1/sqrt(KQ), k = x@Wk + bk  (f16 out, f32 acc)
// ---------------------------------------------------------------------------
__global__ __launch_bounds__(256) void proj_kernel(const _Float16* __restrict__ xh,
                                                   const _Float16* __restrict__ wqt,
                                                   const _Float16* __restrict__ wkt,
                                                   const float* __restrict__ bq,
                                                   const float* __restrict__ bk,
                                                   _Float16* __restrict__ qh,
                                                   _Float16* __restrict__ kh) {
    int wave = threadIdx.x >> 5;
    int lane = threadIdx.x & 31;
    int rowBase = blockIdx.x * 16;
    int nBase   = wave * 16;
    int l16  = lane & 15;
    int koff = (lane >> 4) * 16;     // contiguous k order (A and B alike): hi half-wave holds K 16..31

    const _Float16* aRow  = xh  + (size_t)(rowBase + l16) * D_ + koff;
    const _Float16* bqRow = wqt + (size_t)(nBase  + l16) * D_ + koff;
    const _Float16* bkRow = wkt + (size_t)(nBase  + l16) * D_ + koff;

    v8f cq = {};
    v8f ck = {};
    for (int ch = 0; ch < D_ / 32; ++ch) {
        v16h a  = *(const v16h*)(aRow  + ch * 32);
        v16h bw = *(const v16h*)(bqRow + ch * 32);
        v16h bv = *(const v16h*)(bkRow + ch * 32);
        cq = __builtin_amdgcn_wmma_f32_16x16x32_f16(false, a, false, bw, (short)0, cq, false, false);
        ck = __builtin_amdgcn_wmma_f32_16x16x32_f16(false, a, false, bv, (short)0, ck, false, false);
    }

    float biasq = bq[nBase + l16];
    float biask = bk[nBase + l16];
    const float scale = 0.08838834764831845f;   // 1/sqrt(128)
    int rowOff = (lane >> 4) * 8;               // C layout: lanes>=16 hold rows 8..15
    #pragma unroll
    for (int r = 0; r < 8; ++r) {
        size_t row = (size_t)(rowBase + rowOff + r);
        qh[row * KQ_ + nBase + l16] = (_Float16)((cq[r] + biasq) * scale);
        kh[row * KQ_ + nBase + l16] = (_Float16)(ck[r] + biask);
    }
}

// ---------------------------------------------------------------------------
// TDM: DMA one 64-key x 256B slab of k into LDS with 4-DWORD row padding.
// D# built per CDNA5 ISA §8.3/8.4. data_size=8B: tensor_dim0=32 units,
// dim0_stride=32, tile=32x64, pad_interval=64 DW (enc 5), pad_amount=4 DW
// (enc 3) -> LDS row stride 272B (KSTRIDE halfs).
// ---------------------------------------------------------------------------
#if __has_builtin(__builtin_amdgcn_tensor_load_to_lds)
#define HAVE_TDM 1
__device__ __attribute__((always_inline))
static void tdm_load_slab(const _Float16* gsrc, unsigned int lds_off) {
    unsigned long long ga = (unsigned long long)(size_t)gsrc;
    v4u g0 = { 1u,                                     // [1:0] count=1
               lds_off,                                // [63:32] lds_addr (bytes)
               (unsigned int)(ga & 0xFFFFFFFFu),       // [95:64] global_addr lo
               (unsigned int)(((ga >> 32) & 0x01FFFFFFu) | 0x80000000u) }; // +type=2
    v8i g1 = { (int)((3u << 16) | (1u << 20) | (5u << 22) | (3u << 25)),
               //      data_size=8B  pad_en     pad_int=64DW  pad_amt=4DW
               (int)(32u << 16),                       // tensor_dim0 = 32 (8B units)
               (int)((unsigned)T_ << 16),              // tensor_dim1 = T
               (int)(32u << 16),                       // tile_dim0 = 32
               64,                                     // tile_dim1 = 64 keys
               32,                                     // tensor_dim0_stride = 32
               0, 0 };
    v4i g2 = { 0, 0, 0, 0 };
    v4i g3 = { 0, 0, 0, 0 };
#if defined(__clang_major__) && (__clang_major__ >= 23)
    v8i g4 = { 0, 0, 0, 0, 0, 0, 0, 0 };
    __builtin_amdgcn_tensor_load_to_lds(g0, g1, g2, g3, g4, 0);
#else
    __builtin_amdgcn_tensor_load_to_lds(g0, g1, g2, g3, 0);
#endif
}
#else
#define HAVE_TDM 0
#endif

// ---------------------------------------------------------------------------
// Streaming sim + top-8 + gather. Block = 8 waves owns 32 query rows.
// k slabs are TDM-DMA'd into double-buffered LDS (slab i+1 in flight while
// all 8 waves run WMMAs on slab i). Wave w: rowTile=w/4, colTile=w%4.
// ---------------------------------------------------------------------------
__global__ __launch_bounds__(256) void topk_kernel(const _Float16* __restrict__ qh,
                                                   const _Float16* __restrict__ kh,
                                                   const float* __restrict__ gates,
                                                   const float* __restrict__ am,
                                                   const float* __restrict__ x,
                                                   float* __restrict__ outG,
                                                   float* __restrict__ outIdx,
                                                   float* __restrict__ outVal) {
    __shared__ _Float16 kbuf[2][64 * KSTRIDE];   // 2 x 17KB padded key slabs
    __shared__ float    sc[32][65];              // 64-key score slab, padded
    __shared__ int      idxShare[32][TOPK];

    int wave = threadIdx.x >> 5;
    int lane = threadIdx.x & 31;
    int rowBlock = blockIdx.x * 32;     // global row base (never crosses batch)
    int b = rowBlock / T_;
    int rowTile = wave >> 2;            // 0..1
    int colTile = wave & 3;             // 0..3
    int waveRowBase = rowBlock + rowTile * 16;

    int l16  = lane & 15;
    int koff = (lane >> 4) * 16;
    int rowOff = (lane >> 4) * 8;

    // Preload A = 16 q-rows x KQ=128 (4 chunks, resident across key loop)
    const _Float16* aRow = qh + (size_t)(waveRowBase + l16) * KQ_ + koff;
    v16h a0 = *(const v16h*)(aRow);
    v16h a1 = *(const v16h*)(aRow + 32);
    v16h a2 = *(const v16h*)(aRow + 64);
    v16h a3 = *(const v16h*)(aRow + 96);

    // Per-lane row constants for the 8 rows this lane produces
    float gate_r[8], amr_r[8];
    #pragma unroll
    for (int r = 0; r < 8; ++r) {
        int row = waveRowBase + rowOff + r;
        gate_r[r] = gates[row];
        amr_r[r]  = am[row];
    }

    // Top-8 state (owner threads 0..31, one per row of the 32-row block)
    float tv[TOPK];
    int   ti[TOPK];
    #pragma unroll
    for (int r = 0; r < TOPK; ++r) { tv[r] = -3.4e38f; ti[r] = 0; }
    float minv = -3.4e38f;
    int minSlot = 0;
    int ownRow = threadIdx.x;

    const float*    amB = am + (size_t)b * T_;
    const _Float16* khB = kh + (size_t)b * T_ * KQ_;
    const int nSlabs = T_ / 64;

#if HAVE_TDM
    unsigned int ldsK0 = (unsigned int)(size_t)&kbuf[0][0];
    unsigned int ldsK1 = (unsigned int)(size_t)&kbuf[1][0];
    if (wave == 0) {
        tdm_load_slab(khB, ldsK0);
        __builtin_amdgcn_s_wait_tensorcnt(0);
    }
    __syncthreads();
#endif

    for (int i = 0; i < nSlabs; ++i) {
        int s = i * 64;
#if HAVE_TDM
        if (wave == 0 && i + 1 < nSlabs)
            tdm_load_slab(khB + (size_t)(s + 64) * KQ_, ((i + 1) & 1) ? ldsK1 : ldsK0);
        const _Float16* bRow = &kbuf[i & 1][(colTile * 16 + l16) * KSTRIDE + koff];
#else
        // Fallback: cooperative global->LDS staging of the 64-key slab
        {
            const v4f* src = (const v4f*)(khB + (size_t)s * KQ_);
            int rowsPerPass = 256 / 16;                 // 16 threads per key row
            for (int r0 = 0; r0 < 64; r0 += rowsPerPass) {
                int kr = r0 + (threadIdx.x >> 4);
                int cw = threadIdx.x & 15;              // 16 x 16B = 256B row
                *(v4f*)&kbuf[i & 1][kr * KSTRIDE + cw * 8] =
                    src[(size_t)kr * (KQ_ / 8) + cw];
            }
        }
        __syncthreads();
        const _Float16* bRow = &kbuf[i & 1][(colTile * 16 + l16) * KSTRIDE + koff];
#endif
        v16h b0 = *(const v16h*)(bRow);
        v16h b1 = *(const v16h*)(bRow + 32);
        v16h b2 = *(const v16h*)(bRow + 64);
        v16h b3 = *(const v16h*)(bRow + 96);
        v8f c = {};
        c = __builtin_amdgcn_wmma_f32_16x16x32_f16(false, a0, false, b0, (short)0, c, false, false);
        c = __builtin_amdgcn_wmma_f32_16x16x32_f16(false, a1, false, b1, (short)0, c, false, false);
        c = __builtin_amdgcn_wmma_f32_16x16x32_f16(false, a2, false, b2, (short)0, c, false, false);
        c = __builtin_amdgcn_wmma_f32_16x16x32_f16(false, a3, false, b3, (short)0, c, false, false);

        int keyBase = s + colTile * 16;
        float amc = amB[keyBase + l16];
        #pragma unroll
        for (int r = 0; r < 8; ++r) {
            float v = c[r];
            if (amr_r[r] * amc == 0.0f) v = -1e9f;   // pair mask
            v *= gate_r[r];                           // per-query gate
            sc[rowTile * 16 + rowOff + r][colTile * 16 + l16] = v;
        }
        __syncthreads();

        if (ownRow < 32) {
            for (int j = 0; j < 64; ++j) {
                float v = sc[ownRow][j];
                if (v > minv) {
                    int gidx = s + j;
                    #pragma unroll
                    for (int q = 0; q < TOPK; ++q)
                        if (q == minSlot) { tv[q] = v; ti[q] = gidx; }
                    minv = tv[0]; minSlot = 0;
                    #pragma unroll
                    for (int q = 1; q < TOPK; ++q)
                        if (tv[q] < minv) { minv = tv[q]; minSlot = q; }
                }
            }
        }
#if HAVE_TDM
        if (wave == 0 && i + 1 < nSlabs)
            __builtin_amdgcn_s_wait_tensorcnt(0);
#endif
        __syncthreads();
    }

    // Sort descending (stable bubble; ascending scan kept lowest index on ties)
    if (ownRow < 32) {
        #pragma unroll
        for (int i = 0; i < TOPK; ++i)
            #pragma unroll
            for (int j = 0; j < TOPK - 1 - i; ++j)
                if (tv[j] < tv[j + 1]) {
                    float fv = tv[j]; tv[j] = tv[j + 1]; tv[j + 1] = fv;
                    int   iv = ti[j]; ti[j] = ti[j + 1]; ti[j + 1] = iv;
                }
        size_t grow = (size_t)(rowBlock + ownRow);
        #pragma unroll
        for (int q = 0; q < TOPK; ++q) {
            outIdx[grow * TOPK + q] = (float)ti[q];
            outVal[grow * TOPK + q] = tv[q];
            idxShare[ownRow][q] = ti[q];
        }
    }
    __syncthreads();

    // Gather: 32 rows x 8 picks = 256 (row,k) pairs; each wave copies a pair
    // as 8 steps of 32 lanes x 16B (fully coalesced b128 traffic).
    const float* xB = x + (size_t)b * T_ * D_;
    for (int p = wave; p < 32 * TOPK; p += 8) {
        int rl = p >> 3;
        int kk = p & 7;
        int srcRow = idxShare[rl][kk];
        const v4f* src = (const v4f*)(xB + (size_t)srcRow * D_);
        v4f* dst = (v4f*)(outG + ((size_t)(rowBlock + rl) * TOPK + kk) * D_);
        #pragma unroll
        for (int stp = 0; stp < D_ / 4 / 32; ++stp)
            dst[lane + stp * 32] = src[lane + stp * 32];
    }
}

// ---------------------------------------------------------------------------
extern "C" void kernel_launch(void* const* d_in, const int* in_sizes, int n_in,
                              void* d_out, int out_size, void* d_ws, size_t ws_size,
                              hipStream_t stream) {
    (void)in_sizes; (void)n_in; (void)out_size; (void)ws_size;
    const float* x  = (const float*)d_in[0];
    const float* am = (const float*)d_in[1];
    const float* Wq = (const float*)d_in[2];
    const float* bq = (const float*)d_in[3];
    const float* Wk = (const float*)d_in[4];
    const float* bk = (const float*)d_in[5];
    const float* Wg = (const float*)d_in[6];
    const float* bg = (const float*)d_in[7];

    char* ws = (char*)d_ws;
    size_t off = 0;
    _Float16* xh  = (_Float16*)(ws + off); off += (size_t)B_ * T_ * D_ * 2;
    _Float16* wqt = (_Float16*)(ws + off); off += (size_t)KQ_ * D_ * 2;
    _Float16* wkt = (_Float16*)(ws + off); off += (size_t)KQ_ * D_ * 2;
    _Float16* qh  = (_Float16*)(ws + off); off += (size_t)B_ * T_ * KQ_ * 2;
    _Float16* kh  = (_Float16*)(ws + off); off += (size_t)B_ * T_ * KQ_ * 2;
    float*    gts = (float*)(ws + off);    off += (size_t)B_ * T_ * 4;

    float* outG   = (float*)d_out;                       // [B,T,K,D]
    float* outIdx = outG + (size_t)B_ * T_ * TOPK * D_;  // [B,T,K] (as float)
    float* outVal = outIdx + (size_t)B_ * T_ * TOPK;     // [B,T,K]

    cvt_x_kernel<<<(B_ * T_ * D_) / (256 * 4), 256, 0, stream>>>(x, xh);
    cvt_w_kernel<<<(D_ * KQ_) / 256, 256, 0, stream>>>(Wq, wqt);
    cvt_w_kernel<<<(D_ * KQ_) / 256, 256, 0, stream>>>(Wk, wkt);
    gate_kernel<<<(B_ * T_) / 8, 256, 0, stream>>>(x, Wg, bg, gts);
    proj_kernel<<<(B_ * T_) / 16, 256, 0, stream>>>(xh, wqt, wkt, bq, bk, qh, kh);
    topk_kernel<<<(B_ * T_) / 32, 256, 0, stream>>>(qh, kh, gts, am, x, outG, outIdx, outVal);
}
